// AttnBlock_9225589751966
// MI455X (gfx1250) — matrix operand order repaired and checked by reading, not compiled, not run on hardware
//
#include <hip/hip_runtime.h>
#include <hip/hip_bf16.h>

#define DEVINL __device__ __forceinline__

#define C_   256
#define S_   4096      // pixels per batch image (64*64)
#define B_   2
#define NH_  4
#define D_   64        // head dim
#define NG_  32
#define CPG_ 8         // channels per group

typedef __attribute__((ext_vector_type(16))) __bf16          v16bf;
typedef __attribute__((ext_vector_type(8)))  float           v8f;
typedef __attribute__((ext_vector_type(8)))  unsigned short  v8u16;

union FragU {
    struct { v8u16 lo, hi; } p;
    v16bf v;
};

DEVINL unsigned short f2bf(float f) {
    unsigned int u = __float_as_uint(f);
    unsigned int r = u + 0x7FFFu + ((u >> 16) & 1u);   // round-to-nearest-even
    return (unsigned short)(r >> 16);
}

DEVINL v8f vzero8() {
    v8f z = {0.f, 0.f, 0.f, 0.f, 0.f, 0.f, 0.f, 0.f};
    return z;
}

// ---- WMMA wrapper: D = A(16x32 bf16) * B(32x16 bf16) + C(16x16 f32) ----
DEVINL v8f wmma_bf16(v16bf a, v16bf b, v8f c) {
    return __builtin_amdgcn_wmma_f32_16x16x32_bf16(
        /*neg_a=*/false, a, /*neg_b=*/false, b,
        /*c_mod=*/(short)0, c, /*reuse_a=*/false, /*reuse_b=*/false);
}

// A-matrix fragment (16x32 bf16) from row-major storage mem[row][k], ld elems/row.
// ISA layout: lane m=L%16; lanes 0-15: K = {kb+0..7, kb+16..23}, lanes 16-31: K = {kb+8..15, kb+24..31}
DEVINL v16bf load_frag_a(const unsigned short* base, int ld, int rbase, int kbase, int lane) {
    int m = lane & 15, h = lane >> 4;
    const unsigned short* p = base + (size_t)(rbase + m) * ld + kbase + h * 8;
    FragU f;
    f.p.lo = *(const v8u16*)p;
    f.p.hi = *(const v8u16*)(p + 16);
    return f.v;
}

// B-matrix fragment (32x16 bf16, K x N) from COLUMN-major storage mem[col][row] (row=K contiguous).
// ISA layout: lane n=L%16; lanes 0-15: K = {kb+0..7, kb+16..23}, lanes 16-31: K = {kb+8..15, kb+24..31}
DEVINL v16bf load_frag_b_cm(const unsigned short* base, int ldcol, int nbase, int kbase, int lane) {
    int n = lane & 15, h = lane >> 4;
    const unsigned short* p = base + (size_t)(nbase + n) * ldcol + kbase + h * 8;
    FragU f;
    f.p.lo = *(const v8u16*)p;
    f.p.hi = *(const v8u16*)(p + 16);
    return f.v;
}

// ---------------- 1) fp32 -> bf16 weight conversion (wq,wk,wv,wp) ----------------
__global__ void wcvt_kernel(const float* __restrict__ wq, const float* __restrict__ wk,
                            const float* __restrict__ wv, const float* __restrict__ wp,
                            unsigned short* __restrict__ dst) {
    int idx = blockIdx.x * 256 + threadIdx.x;          // 0 .. 4*65536-1
    int which = idx >> 16;
    int off   = idx & 65535;
    const float* src = (which == 0) ? wq : (which == 1) ? wk : (which == 2) ? wv : wp;
    dst[idx] = f2bf(src[off]);
}

// ---------------- 2) GroupNorm -> hn bf16, layout [pixel_global][channel] ----------------
__global__ void gn_kernel(const float* __restrict__ x, const float* __restrict__ gamma,
                          const float* __restrict__ beta, unsigned short* __restrict__ hn) {
    __shared__ float ssum[256], ssq[256];
    int b = blockIdx.x >> 5;       // 32 groups per batch
    int g = blockIdx.x & 31;
    int t = threadIdx.x;
    const float* xb = x + ((size_t)b * C_ + g * CPG_) * S_;
    float s = 0.f, q = 0.f;
    for (int i = t; i < CPG_ * S_; i += 256) {
        float v = xb[i];
        s += v; q += v * v;
    }
    ssum[t] = s; ssq[t] = q;
    __syncthreads();
    for (int off = 128; off > 0; off >>= 1) {
        if (t < off) { ssum[t] += ssum[t + off]; ssq[t] += ssq[t + off]; }
        __syncthreads();
    }
    const float inv = 1.0f / (float)(CPG_ * S_);
    float mu  = ssum[0] * inv;
    float var = ssq[0] * inv - mu * mu;
    float rs  = rsqrtf(var + 1e-5f);
    for (int i = t; i < CPG_ * S_; i += 256) {
        int cl = i / S_, sp = i - cl * S_;
        int c  = g * CPG_ + cl;
        float v = (xb[i] - mu) * rs * gamma[c] + beta[c];
        hn[((size_t)b * S_ + sp) * C_ + c] = f2bf(v);
    }
}

// ---------------- 3) QKV projection: wave computes 16(o) x 16(pix) for q,k,v ----------------
// q,k stored [b][n][s][d]; v stored [b][n][d][s]
__global__ void qkv_kernel(const unsigned short* __restrict__ wbf,
                           const float* __restrict__ bq, const float* __restrict__ bk,
                           const float* __restrict__ bv,
                           const unsigned short* __restrict__ hn,
                           unsigned short* __restrict__ qo,
                           unsigned short* __restrict__ ko,
                           unsigned short* __restrict__ vo) {
    int lane = threadIdx.x & 31;
    int w    = blockIdx.x * 8 + (threadIdx.x >> 5);
    int ot = w & 15;               // 16 output-channel tiles
    int pt = w >> 4;               // 512 pixel tiles (B*S/16)
    int obase = ot * 16;
    int pbase = pt * 16;
    const unsigned short* wq = wbf;
    const unsigned short* wk = wbf + C_ * C_;
    const unsigned short* wv = wbf + 2 * C_ * C_;
    v8f accq = vzero8(), acck = vzero8(), accv = vzero8();
    for (int cb = 0; cb < C_; cb += 32) {
        v16bf bfr = load_frag_b_cm(hn, C_, pbase, cb, lane);   // hn column-major in (c,pix)
        v16bf aq  = load_frag_a(wq, C_, obase, cb, lane);
        v16bf ak  = load_frag_a(wk, C_, obase, cb, lane);
        v16bf av  = load_frag_a(wv, C_, obase, cb, lane);
        accq = wmma_bf16(aq, bfr, accq);
        acck = wmma_bf16(ak, bfr, acck);
        accv = wmma_bf16(av, bfr, accv);
    }
    int nl = lane & 15, hi = lane >> 4;
    int pix = pbase + nl;          // global pixel index across batch
    int b = pix >> 12, sp = pix & (S_ - 1);
    for (int r = 0; r < 8; ++r) {
        int o  = obase + r + hi * 8;
        int d  = o >> 2;           // c = d*NH + n  ->  d = o/4
        int nh = o & 3;
        float fq = accq[r] + bq[o];
        float fk = acck[r] + bk[o];
        float fv = accv[r] + bv[o];
        size_t qidx = (((size_t)(b * NH_ + nh) * S_) + sp) * D_ + d;
        qo[qidx] = f2bf(fq);
        ko[qidx] = f2bf(fk);
        vo[(((size_t)(b * NH_ + nh) * D_) + d) * S_ + sp] = f2bf(fv);
    }
}

// ---------------- 4) Flash attention: 1 wave per (b, head, 16-query tile) ----------------
__global__ void attn_kernel(const unsigned short* __restrict__ q,
                            const unsigned short* __restrict__ k,
                            const unsigned short* __restrict__ v,
                            unsigned short* __restrict__ ao) {
    __shared__ __align__(16) unsigned short pbuf[8][16 * 32];   // per-wave P staging
    int lane = threadIdx.x & 31;
    int wid  = threadIdx.x >> 5;
    int w  = blockIdx.x * 8 + wid;         // 0 .. 2047
    int qt = w & 255;
    int n  = (w >> 8) & 3;
    int b  = w >> 10;
    int qbase = qt * 16;
    const unsigned short* qh = q + (size_t)(b * NH_ + n) * S_ * D_;   // [s][d]
    const unsigned short* kh = k + (size_t)(b * NH_ + n) * S_ * D_;   // [s][d]
    const unsigned short* vh = v + (size_t)(b * NH_ + n) * D_ * S_;   // [d][s]

    v16bf aq0 = load_frag_a(qh, D_, qbase, 0,  lane);   // Q tile, d = 0..31
    v16bf aq1 = load_frag_a(qh, D_, qbase, 32, lane);   // Q tile, d = 32..63

    v8f oacc[4];
    float mrow[8], lrow[8];
    for (int t = 0; t < 4; ++t) oacc[t] = vzero8();
    for (int r = 0; r < 8; ++r) { mrow[r] = -1e30f; lrow[r] = 0.f; }

    unsigned short* myp = &pbuf[wid][0];
    const float scale = 0.125f;            // 64^-0.5

    for (int ks = 0; ks < S_; ks += 32) {
        if (ks + 32 < S_) {
            __builtin_prefetch(kh + (size_t)(ks + 32) * D_, 0, 1);     // global_prefetch_b8
            __builtin_prefetch(vh + (size_t)lane * S_ + ks + 32, 0, 1);
        }
        // ---- scores for two 16-key subtiles (K over d=64 in two steps) ----
        v8f s0 = vzero8(), s1 = vzero8();
        {
            v16bf bk0 = load_frag_b_cm(kh, D_, ks,      0,  lane);
            v16bf bk1 = load_frag_b_cm(kh, D_, ks,      32, lane);
            s0 = wmma_bf16(aq0, bk0, s0);
            s0 = wmma_bf16(aq1, bk1, s0);
            v16bf bk2 = load_frag_b_cm(kh, D_, ks + 16, 0,  lane);
            v16bf bk3 = load_frag_b_cm(kh, D_, ks + 16, 32, lane);
            s1 = wmma_bf16(aq0, bk2, s1);
            s1 = wmma_bf16(aq1, bk3, s1);
        }
        // ---- online softmax over 32 keys; rows live in half-wave 16-lane groups ----
        float corr[8];
        for (int r = 0; r < 8; ++r) {
            float a = s0[r] * scale, c2 = s1[r] * scale;
            float mx = fmaxf(a, c2);
            mx = fmaxf(mx, __shfl_xor(mx, 1));
            mx = fmaxf(mx, __shfl_xor(mx, 2));
            mx = fmaxf(mx, __shfl_xor(mx, 4));
            mx = fmaxf(mx, __shfl_xor(mx, 8));
            float mnew = fmaxf(mrow[r], mx);
            float cr   = __expf(mrow[r] - mnew);
            float p0   = __expf(a  - mnew);
            float p1   = __expf(c2 - mnew);
            s0[r] = p0; s1[r] = p1;
            float rs = p0 + p1;
            rs += __shfl_xor(rs, 1);
            rs += __shfl_xor(rs, 2);
            rs += __shfl_xor(rs, 4);
            rs += __shfl_xor(rs, 8);
            lrow[r] = lrow[r] * cr + rs;
            mrow[r] = mnew;
            corr[r] = cr;
        }
        for (int t = 0; t < 4; ++t)
            for (int r = 0; r < 8; ++r) oacc[t][r] *= corr[r];
        // ---- D-layout P -> LDS (row-major 16x32 bf16) ----
        {
            int nl = lane & 15, hi = lane >> 4;
            for (int r = 0; r < 8; ++r) {
                int row = r + hi * 8;
                myp[row * 32 + nl]      = f2bf(s0[r]);
                myp[row * 32 + 16 + nl] = f2bf(s1[r]);
            }
        }
        // ---- LDS -> A fragment (16q x 32k) ----
        v16bf pa;
        {
            int m = lane & 15, hi = lane >> 4;
            const unsigned short* pp = myp + m * 32 + hi * 8;
            FragU f;
            f.p.lo = *(const v8u16*)pp;
            f.p.hi = *(const v8u16*)(pp + 16);
            pa = f.v;
        }
        // ---- O += P * V^T : four 16-d column tiles ----
        for (int t = 0; t < 4; ++t) {
            v16bf bvf = load_frag_b_cm(vh, S_, t * 16, ks, lane);   // col=d, rows=key pixels
            oacc[t] = wmma_bf16(pa, bvf, oacc[t]);
        }
    }
    // ---- epilogue: O /= l, store bf16 to ao[b][pixel][c], c = d*NH + n ----
    int nl = lane & 15, hi = lane >> 4;
    for (int t = 0; t < 4; ++t) {
        int d = t * 16 + nl;
        int c = d * NH_ + n;
        for (int r = 0; r < 8; ++r) {
            int pix = qbase + r + hi * 8;
            float val = oacc[t][r] / lrow[r];
            ao[((size_t)b * S_ + pix) * C_ + c] = f2bf(val);
        }
    }
}

// ---------------- 5) output projection + bias + residual ----------------
__global__ void proj_kernel(const unsigned short* __restrict__ wp,
                            const float* __restrict__ bp,
                            const unsigned short* __restrict__ ao,
                            const float* __restrict__ x,
                            float* __restrict__ out) {
    int lane = threadIdx.x & 31;
    int w    = blockIdx.x * 8 + (threadIdx.x >> 5);
    int ot = w & 15, pt = w >> 4;
    int obase = ot * 16, pbase = pt * 16;
    v8f acc = vzero8();
    for (int cb = 0; cb < C_; cb += 32) {
        v16bf a   = load_frag_a(wp, C_, obase, cb, lane);
        v16bf bfr = load_frag_b_cm(ao, C_, pbase, cb, lane);
        acc = wmma_bf16(a, bfr, acc);
    }
    int nl = lane & 15, hi = lane >> 4;
    int pix = pbase + nl;
    int b = pix >> 12, sp = pix & (S_ - 1);
    for (int r = 0; r < 8; ++r) {
        int o = obase + r + hi * 8;
        size_t idx = ((size_t)(b * C_ + o)) * S_ + sp;
        out[idx] = x[idx] + acc[r] + bp[o];
    }
}

extern "C" void kernel_launch(void* const* d_in, const int* in_sizes, int n_in,
                              void* d_out, int out_size, void* d_ws, size_t ws_size,
                              hipStream_t stream) {
    (void)in_sizes; (void)n_in; (void)out_size; (void)ws_size;
    const float* x     = (const float*)d_in[0];
    const float* gamma = (const float*)d_in[1];
    const float* beta  = (const float*)d_in[2];
    const float* wq    = (const float*)d_in[3];
    const float* bq    = (const float*)d_in[4];
    const float* wk    = (const float*)d_in[5];
    const float* bk    = (const float*)d_in[6];
    const float* wv    = (const float*)d_in[7];
    const float* bv    = (const float*)d_in[8];
    const float* wp    = (const float*)d_in[9];
    const float* bp    = (const float*)d_in[10];
    float* out = (float*)d_out;

    // workspace layout (bf16 = unsigned short elements)
    unsigned short* ws  = (unsigned short*)d_ws;
    unsigned short* hn  = ws;                         // [B*S][C]        = 2,097,152
    unsigned short* wbf = hn  + (size_t)B_ * S_ * C_; // 4 x [C][C]      =   262,144
    unsigned short* qb  = wbf + 4 * C_ * C_;          // [B][NH][S][D]   = 2,097,152
    unsigned short* kb  = qb  + (size_t)B_ * NH_ * S_ * D_;
    unsigned short* vb  = kb  + (size_t)B_ * NH_ * S_ * D_;   // [B][NH][D][S]
    unsigned short* ao  = vb  + (size_t)B_ * NH_ * S_ * D_;   // [B*S][C]

    wcvt_kernel<<<1024, 256, 0, stream>>>(wq, wk, wv, wp, wbf);
    gn_kernel  <<<B_ * NG_, 256, 0, stream>>>(x, gamma, beta, hn);
    qkv_kernel <<<1024, 256, 0, stream>>>(wbf, bq, bk, bv, hn, qb, kb, vb);
    attn_kernel<<<256, 256, 0, stream>>>(qb, kb, vb, ao);
    proj_kernel<<<1024, 256, 0, stream>>>(wbf + 3 * C_ * C_, bp, ao, x, out);
}
